// EvoformerBlock_79602923864162
// MI455X (gfx1250) — matrix operand order repaired and checked by reading, not compiled, not run on hardware
//
#include <hip/hip_runtime.h>

// ---------------------------------------------------------------------------
// Evoformer block for gfx1250 (MI455X). All contractions -> v_wmma_f32_16x16x32_f16.
// wave32. Fragment layouts per CDNA5 ISA:
//   A frag (16x32 f16): lane m=l&15; elem i -> k = (i>>3)*16 + (l>>4)*8 + (i&7)
//     => per lane: two contiguous 8-float runs at k=hw*8 and k=16+hw*8
//   B frag (32x16 f16): lane n=l&15; elem i -> k = (i>>3)*16 + (l>>4)*8 + (i&7)
//     => per lane: two contiguous 8-float runs at k=hw*8 and k=16+hw*8
//   C/D  (16x16 f32):   lane: n=l&15; vgpr r -> m = r + 8*(l>>4)
// ---------------------------------------------------------------------------

typedef __attribute__((ext_vector_type(16))) _Float16 v16h;
typedef __attribute__((ext_vector_type(8)))  _Float16 v8h;
typedef __attribute__((ext_vector_type(8)))  float    v8f;
typedef __attribute__((ext_vector_type(4)))  float    v4f;

#define N_SEQ  128
#define N_RES  256
#define C_MSA  64
#define C_PAIR 128
#define SR (N_SEQ * N_RES)   // 32768 msa rows (s,i)
#define MR (N_RES * N_RES)   // 65536 pair rows (i,j)

__device__ __forceinline__ v8f wmma_f16(v16h a, v16h b, v8f c) {
  return __builtin_amdgcn_wmma_f32_16x16x32_f16(false, a, false, b, (short)0, c,
                                                false, false);
}
__device__ __forceinline__ int ka_off(int hw, int i) { return ((i >> 3) << 4) | (hw << 3) | (i & 7); }
__device__ __forceinline__ int kb_off(int hw, int i) { return ((i >> 3) << 4) | (hw << 3) | (i & 7); }
__device__ __forceinline__ float sigm(float x) { return 1.f / (1.f + __expf(-x)); }

__device__ __forceinline__ v16h pack_a(v4f x0, v4f x1, v4f x2, v4f x3) {
  v16h f;
#pragma unroll
  for (int i = 0; i < 4; ++i) {
    f[i] = (_Float16)x0[i];
    f[4 + i] = (_Float16)x1[i];
    f[8 + i] = (_Float16)x2[i];
    f[12 + i] = (_Float16)x3[i];
  }
  return f;
}

// ---------------------------------------------------------------------------
// Generic strided/batched WMMA GEMM: D[by,bz][m,n] (+)= alpha * sum_k A[m,k]*B[k,n]
// GUARD: bounds checks (only for ragged N). AK1/BK1: k-stride==1 -> b128 loads.
// NW: 16-wide n subtiles per wave (A fragment reuse).
// ---------------------------------------------------------------------------
enum { EPI_STORE = 0, EPI_ADD = 1, EPI_SIGMUL = 2, EPI_ROWMASK = 3 };

template <bool GUARD, bool AK1, bool BK1, int NW>
__global__ __launch_bounds__(32) void k_gemm(
    const float* __restrict__ A, const float* __restrict__ B,
    float* __restrict__ D, const float* __restrict__ E,
    int M, int N, int K, int ntiles,
    long long sAm, long long sAk, long long sAy, long long sAz,
    long long sBk, long long sBn, long long sBy, long long sBz,
    long long sDm, long long sDn, long long sDy, long long sDz,
    float alpha, int epi, long long sEm, long long sEz) {
  const int lane = threadIdx.x;
  const int mt = blockIdx.x / ntiles, nt = blockIdx.x % ntiles;
  const int lm = lane & 15, hw = lane >> 4;
  const int m0 = mt * 16, n0 = nt * 16 * NW;
  const long long aB = (long long)blockIdx.y * sAy + (long long)blockIdx.z * sAz;
  const long long bB = (long long)blockIdx.y * sBy + (long long)blockIdx.z * sBz;
  const long long dB = (long long)blockIdx.y * sDy + (long long)blockIdx.z * sDz;

  const float* Arow = A + aB + (long long)(m0 + lm) * sAm;
  const float* Bcol[NW];
#pragma unroll
  for (int w = 0; w < NW; ++w)
    Bcol[w] = B + bB + (long long)(n0 + w * 16 + lm) * sBn;

  v8f acc[NW] = {};

  for (int kk = 0; kk < K; kk += 32) {
    v16h af;
    if constexpr (!GUARD && AK1) {
      const v4f* p0 = (const v4f*)(Arow + kk + hw * 8);
      const v4f* p1 = (const v4f*)(Arow + kk + 16 + hw * 8);
      af = pack_a(p0[0], p0[1], p1[0], p1[1]);
    } else {
#pragma unroll
      for (int i = 0; i < 16; ++i) {
        int ka = kk + ka_off(hw, i);
        float va;
        if constexpr (GUARD)
          va = (m0 + lm < M && ka < K)
                   ? A[aB + (long long)(m0 + lm) * sAm + (long long)ka * sAk]
                   : 0.f;
        else
          va = Arow[(long long)ka * sAk];
        af[i] = (_Float16)va;
      }
    }
#pragma unroll
    for (int w = 0; w < NW; ++w) {
      v16h bf;
      if constexpr (!GUARD && BK1) {
        const v4f* q0 = (const v4f*)(Bcol[w] + kk + hw * 8);
        const v4f* q1 = (const v4f*)(Bcol[w] + kk + 16 + hw * 8);
        bf = pack_a(q0[0], q0[1], q1[0], q1[1]);
      } else {
#pragma unroll
        for (int i = 0; i < 16; ++i) {
          int kb = kk + kb_off(hw, i);
          float vb;
          if constexpr (GUARD)
            vb = (n0 + w * 16 + lm < N && kb < K)
                     ? B[bB + (long long)kb * sBk + (long long)(n0 + w * 16 + lm) * sBn]
                     : 0.f;
          else
            vb = Bcol[w][(long long)kb * sBk];
          bf[i] = (_Float16)vb;
        }
      }
      acc[w] = wmma_f16(af, bf, acc[w]);
    }
  }
#pragma unroll
  for (int w = 0; w < NW; ++w) {
#pragma unroll
    for (int r = 0; r < 8; ++r) {
      int m = m0 + r + (hw << 3);
      int n = n0 + w * 16 + lm;
      if (GUARD && (m >= M || n >= N)) continue;
      float val = acc[w][r] * alpha;
      long long di = dB + (long long)m * sDm + (long long)n * sDn;
      if (epi == EPI_STORE)       D[di] = val;
      else if (epi == EPI_ADD)    D[di] += val;
      else if (epi == EPI_SIGMUL) D[di] *= sigm(val);
      else                        D[di] = val * E[(long long)m * sEm + (long long)blockIdx.z * sEz];
    }
  }
}

// ---------------------------------------------------------------------------
// LayerNorm: one wave per row, C in {64,128}
// ---------------------------------------------------------------------------
__global__ __launch_bounds__(32) void k_ln(const float* __restrict__ x,
                                           const float* __restrict__ g,
                                           const float* __restrict__ b,
                                           float* __restrict__ y, int C) {
  long long row = blockIdx.x;
  int lane = threadIdx.x;
  int np = C >> 5;
  float v[4];
  float s = 0.f;
  for (int t = 0; t < np; ++t) {
    v[t] = x[row * C + lane + 32 * t];
    s += v[t];
  }
  for (int off = 16; off; off >>= 1) s += __shfl_xor(s, off);
  float mean = s / (float)C;
  float q = 0.f;
  for (int t = 0; t < np; ++t) { float d = v[t] - mean; q += d * d; }
  for (int off = 16; off; off >>= 1) q += __shfl_xor(q, off);
  float inv = rsqrtf(q / (float)C + 1e-5f);
  for (int t = 0; t < np; ++t) {
    int c = lane + 32 * t;
    y[row * C + c] = (v[t] - mean) * inv * g[c] + b[c];
  }
}

// Softmax over rows of length 256 with optional column mask bias 1e9*(cm[c]-1)
__global__ __launch_bounds__(32) void k_softmax256(float* __restrict__ X,
                                                   const float* __restrict__ cm) {
  long long row = blockIdx.x;
  int lane = threadIdx.x;
  float v[8];
#pragma unroll
  for (int t = 0; t < 8; ++t) {
    int c = lane + 32 * t;
    v[t] = X[row * 256 + c] + (cm ? 1e9f * (cm[c] - 1.f) : 0.f);
  }
  float mx = v[0];
#pragma unroll
  for (int t = 1; t < 8; ++t) mx = fmaxf(mx, v[t]);
  for (int off = 16; off; off >>= 1) mx = fmaxf(mx, __shfl_xor(mx, off));
  float s = 0.f;
#pragma unroll
  for (int t = 0; t < 8; ++t) { v[t] = __expf(v[t] - mx); s += v[t]; }
  for (int off = 16; off; off >>= 1) s += __shfl_xor(s, off);
  float inv = 1.f / s;
#pragma unroll
  for (int t = 0; t < 8; ++t) X[row * 256 + lane + 32 * t] = v[t] * inv;
}

__global__ void k_colmask(const float* __restrict__ mm, float* __restrict__ cm) {
  int j = threadIdx.x;
  float m = 0.f;
  for (int s = 0; s < N_SEQ; ++s) m = fmaxf(m, mm[s * N_RES + j]);
  cm[j] = m;
}
__global__ void k_normij(const float* __restrict__ mm, float* __restrict__ nrm) {
  int i = blockIdx.x, j = threadIdx.x;
  float s = 0.f;
  for (int t = 0; t < N_SEQ; ++t) s += mm[t * N_RES + i] * mm[t * N_RES + j];
  nrm[i * N_RES + j] = s;
}
__global__ void k_glu(const float* __restrict__ H, float* __restrict__ G,
                      long long total, int Hh) {
  long long idx = (long long)blockIdx.x * blockDim.x + threadIdx.x;
  if (idx >= total) return;
  long long row = idx / Hh;
  int h = (int)(idx % Hh);
  float a = H[row * (2LL * Hh) + h];
  float b = H[row * (2LL * Hh) + Hh + h];
  G[idx] = a * sigm(a) * b;  // silu(a)*b
}
__global__ void k_add(float* __restrict__ d, const float* __restrict__ s, long long n) {
  long long idx = (long long)blockIdx.x * blockDim.x + threadIdx.x;
  if (idx < n) d[idx] += s[idx];
}

// ---------------------------------------------------------------------------
// Fused OuterProductMean: per (i, 16-wide j tile).
// Inputs are s-major transposed projections: aT[i][c][s], bT[j][e][s] (s=0..127).
// Phase1: S[(jj,e),c] = sum_s bT[j0+jj][e][s]*aT[i][c][s]  (WMMA -> LDS f16)
// Phase2: out[jj,f]   = Outer[jj,(c,e)] @ W[(c,e),f]       (WMMA from LDS)
// Epilogue: pair[i,j,f] += (out+bias[f])/(1e-3+norm[i,j])
// ---------------------------------------------------------------------------
__global__ __launch_bounds__(256) void k_opm(
    const float* __restrict__ aT, const float* __restrict__ bT,
    const float* __restrict__ W, const float* __restrict__ bias,
    const float* __restrict__ nrm, float* __restrict__ pair) {
  __shared__ __align__(16) _Float16 S[512 * 32];  // [(jj*32+e)][c]
  const int i = blockIdx.y;
  const int j0 = blockIdx.x * 16;
  const int tid = threadIdx.x, wave = tid >> 5, lane = tid & 31;
  const int lm = lane & 15, hw = lane >> 4;

  // phase 1: M=512 (32 mtiles), N=32 (2 ntiles), K=128 (s)
  for (int t = wave; t < 64; t += 8) {
    int mt = t >> 1, ntt = t & 1;
    int m0 = mt * 16, n0 = ntt * 16;
    int mrow = m0 + lm, jj = mrow >> 5, e = mrow & 31;
    const float* Ar = bT + (long long)(j0 + jj) * 4096 + e * 128;
    const float* Br = aT + (long long)i * 4096 + (n0 + lm) * 128;
    v8f acc = {};
    for (int kk = 0; kk < 128; kk += 32) {
      const v4f* p0 = (const v4f*)(Ar + kk + hw * 8);
      const v4f* p1 = (const v4f*)(Ar + kk + 16 + hw * 8);
      v16h af = pack_a(p0[0], p0[1], p1[0], p1[1]);
      const v4f* q0 = (const v4f*)(Br + kk + hw * 8);
      const v4f* q1 = (const v4f*)(Br + kk + 16 + hw * 8);
      v16h bf = pack_a(q0[0], q0[1], q1[0], q1[1]);
      acc = wmma_f16(af, bf, acc);
    }
#pragma unroll
    for (int r = 0; r < 8; ++r) S[(m0 + r + (hw << 3)) * 32 + (n0 + lm)] = (_Float16)acc[r];
  }
  __syncthreads();

  // phase 2: M=16, N=128 (wave -> ntile), K=1024 ((c,e) with k=c*32+e)
  {
    int n0 = wave * 16;
    v8f acc = {};
    for (int kk = 0; kk < 1024; kk += 32) {
      v16h af, bf;
#pragma unroll
      for (int q = 0; q < 16; ++q) {
        int ka = kk + ka_off(hw, q);
        int c = ka >> 5, e = ka & 31;
        af[q] = S[(lm * 32 + e) * 32 + c];
        int kb = kk + kb_off(hw, q);
        bf[q] = (_Float16)W[(long long)kb * C_PAIR + (n0 + lm)];
      }
      acc = wmma_f16(af, bf, acc);
    }
#pragma unroll
    for (int r = 0; r < 8; ++r) {
      int jj = r + (hw << 3);
      int f = n0 + lm;
      int j = j0 + jj;
      long long pidx = ((long long)i * N_RES + j) * C_PAIR + f;
      float v = acc[r] + bias[f];
      pair[pidx] += v / (1e-3f + nrm[i * N_RES + j]);
    }
  }
}

// ---------------------------------------------------------------------------
// Fused grid self-attention (flash-style): per (b, h, 16-wide q tile).
// logits (16x256) via WMMA -> LDS, softmax, P@V via WMMA. d=32, heads=4.
// Transposed variant handled purely via sRow/sB strides (channel stride is 1
// in both layouts so Q/K fragment loads stay b128-vectorized).
// ---------------------------------------------------------------------------
__global__ __launch_bounds__(64) void k_attn(
    const float* __restrict__ q, const float* __restrict__ kbuf,
    const float* __restrict__ vbuf, const float* __restrict__ bias,
    const float* __restrict__ mask, float* __restrict__ o,
    long long sRow, long long sB, long long sORow, long long sOB) {
  __shared__ __align__(16) float LG[16 * 256];
  __shared__ __align__(16) _Float16 PR[16 * 256];
  const int q0 = blockIdx.x * 16;
  const int h = blockIdx.y;
  const int bidx = blockIdx.z;
  const int tid = threadIdx.x, wave = tid >> 5, lane = tid & 31;
  const int lm = lane & 15, hw = lane >> 4;
  const long long base = (long long)bidx * sB + h * 32;

  // phase 1: 16 n-tiles over k-dim 256; K = d = 32 (one wmma each); A reused
  v16h af;
  {
    const float* Aq = q + base + (long long)(q0 + lm) * sRow;
    const v4f* p0 = (const v4f*)(Aq + hw * 8);
    const v4f* p1 = (const v4f*)(Aq + 16 + hw * 8);
    af = pack_a(p0[0], p0[1], p1[0], p1[1]);
  }
  for (int t = wave; t < 16; t += 2) {
    int n0 = t * 16;
    const float* Bk = kbuf + base + (long long)(n0 + lm) * sRow;
    const v4f* q4 = (const v4f*)(Bk + hw * 8);
    const v4f* q5 = (const v4f*)(Bk + 16 + hw * 8);
    v16h bf = pack_a(q4[0], q4[1], q5[0], q5[1]);
    v8f acc = {};
    acc = wmma_f16(af, bf, acc);
#pragma unroll
    for (int r = 0; r < 8; ++r) {
      int m = r + (hw << 3);
      int n = n0 + lm;
      LG[m * 256 + n] = acc[r] +
                        bias[(long long)h * MR + (long long)(q0 + m) * N_RES + n] +
                        1e9f * (mask[bidx * N_RES + n] - 1.f);
    }
  }
  __syncthreads();
  if (tid < 16) {
    float mx = -1e30f;
    for (int c = 0; c < 256; ++c) mx = fmaxf(mx, LG[tid * 256 + c]);
    float s = 0.f;
    for (int c = 0; c < 256; ++c) { float e = __expf(LG[tid * 256 + c] - mx); s += e; LG[tid * 256 + c] = e; }
    float inv = 1.f / s;
    for (int c = 0; c < 256; ++c) PR[tid * 256 + c] = (_Float16)(LG[tid * 256 + c] * inv);
  }
  __syncthreads();

  // phase 3: out 16x32 = P(16x256) @ V(256x32); wave -> n-tile
  {
    int n0 = wave * 16;
    const _Float16* Pr = PR + lm * 256;
    v8f acc = {};
    for (int kk = 0; kk < 256; kk += 32) {
      v16h pf;
      v8h a0 = *(const v8h*)(Pr + kk + hw * 8);
      v8h a1 = *(const v8h*)(Pr + kk + 16 + hw * 8);
#pragma unroll
      for (int i = 0; i < 8; ++i) { pf[i] = a0[i]; pf[8 + i] = a1[i]; }
      v16h bf;
#pragma unroll
      for (int i = 0; i < 16; ++i) {
        int kb = kk + kb_off(hw, i);
        bf[i] = (_Float16)vbuf[base + (long long)kb * sRow + (n0 + lm)];
      }
      acc = wmma_f16(pf, bf, acc);
    }
#pragma unroll
    for (int r = 0; r < 8; ++r) {
      int m = r + (hw << 3);
      o[(long long)bidx * sOB + (long long)(q0 + m) * sORow + h * 32 + (n0 + lm)] = acc[r];
    }
  }
}

// ---------------------------------------------------------------------------
// Host side
// ---------------------------------------------------------------------------
enum {
  IN_MSA = 0, IN_PAIR, IN_MSA_MASK, IN_PAIR_MASK,
  P_OPM_LN_G, P_OPM_LN_B, P_MA_ACT_LN_G, P_MA_ACT_LN_B,
  P_MA_PAIR_LN_G, P_MA_PAIR_LN_B, P_MT_LN_G, P_MT_LN_B,
  P_TMO_LN_G, P_TMO_LN_B, P_TMO_CLN_G, P_TMO_CLN_B,
  P_TMI_LN_G, P_TMI_LN_B, P_TMI_CLN_G, P_TMI_CLN_B,
  P_PA1_LN_G, P_PA1_LN_B, P_PA2_LN_G, P_PA2_LN_B,
  P_PT_LN_G, P_PT_LN_B,
  P_OPM_LEFT_W, P_OPM_RIGHT_W, P_OPM_OUT_W, P_OPM_OUT_B,
  P_MA_PAIR_LOGITS_W, P_MA_V_W, P_MA_GATING_W, P_MA_OUT_W,
  P_MT_W1, P_MT_W2,
  P_TMO_PROJ_W, P_TMO_GATE_W, P_TMO_OUT_W, P_TMO_GATING_W,
  P_TMI_PROJ_W, P_TMI_GATE_W, P_TMI_OUT_W, P_TMI_GATING_W,
  P_PA1_BIAS_W, P_PA1_Q_W, P_PA1_K_W, P_PA1_V_W, P_PA1_GATING_W, P_PA1_OUT_W,
  P_PA2_BIAS_W, P_PA2_Q_W, P_PA2_K_W, P_PA2_V_W, P_PA2_GATING_W, P_PA2_OUT_W,
  P_PT_W1, P_PT_W2
};

static inline void gemm(hipStream_t st, const float* A, const float* B, float* D,
                        const float* E, int M, int N, int K, int by, int bz,
                        long long sAm, long long sAk, long long sAy, long long sAz,
                        long long sBk, long long sBn, long long sBy, long long sBz,
                        long long sDm, long long sDn, long long sDy, long long sDz,
                        float alpha, int epi, long long sEm = 1, long long sEz = 0) {
  bool guard = (M % 16) || (N % 32) || (K % 32);
  if (guard) {
    int nt = (N + 15) / 16;
    dim3 g(((M + 15) / 16) * nt, by, bz);
    k_gemm<true, false, false, 1><<<g, 32, 0, st>>>(
        A, B, D, E, M, N, K, nt, sAm, sAk, sAy, sAz, sBk, sBn, sBy, sBz,
        sDm, sDn, sDy, sDz, alpha, epi, sEm, sEz);
  } else {
    int nt = N / 32;
    dim3 g((M / 16) * nt, by, bz);
    if (sAk == 1 && sBk == 1)
      k_gemm<false, true, true, 2><<<g, 32, 0, st>>>(
          A, B, D, E, M, N, K, nt, sAm, sAk, sAy, sAz, sBk, sBn, sBy, sBz,
          sDm, sDn, sDy, sDz, alpha, epi, sEm, sEz);
    else if (sAk == 1)
      k_gemm<false, true, false, 2><<<g, 32, 0, st>>>(
          A, B, D, E, M, N, K, nt, sAm, sAk, sAy, sAz, sBk, sBn, sBy, sBz,
          sDm, sDn, sDy, sDz, alpha, epi, sEm, sEz);
    else
      k_gemm<false, false, false, 2><<<g, 32, 0, st>>>(
          A, B, D, E, M, N, K, nt, sAm, sAk, sAy, sAz, sBk, sBn, sBy, sBz,
          sDm, sDn, sDy, sDz, alpha, epi, sEm, sEz);
  }
}
// Plain row-major: A[M,K] @ B[K,N] -> D[M,N]
static inline void gemm_rm(hipStream_t st, const float* A, const float* B, float* D,
                           const float* E, int M, int N, int K, float alpha, int epi) {
  gemm(st, A, B, D, E, M, N, K, 1, 1,
       K, 1, 0, 0, N, 1, 0, 0, N, 1, 0, 0, alpha, epi);
}

extern "C" void kernel_launch(void* const* d_in, const int* in_sizes, int n_in,
                              void* d_out, int out_size, void* d_ws, size_t ws_size,
                              hipStream_t stream) {
  (void)in_sizes; (void)n_in; (void)out_size; (void)ws_size;
  auto P = [&](int i) { return (const float*)d_in[i]; };
  const float* msa_mask = P(IN_MSA_MASK);
  const float* pair_mask = P(IN_PAIR_MASK);

  float* out = (float*)d_out;
  float* msaO = out;                           // [SR, 64]
  float* pairO = out + (long long)SR * C_MSA;  // [MR, 128]

  // workspace layout (f32 elems)
  float* W = (float*)d_ws;
  float* BUF0 = W;                        // 16,777,216
  float* BUF1 = BUF0 + 16777216;          //  8,388,608
  float* BUF2 = BUF1 + 8388608;           //  8,388,608 (lnP)
  float* BUF3 = BUF2 + 8388608;           //  2,097,152 (lnM)
  float* BUF4 = BUF3 + 2097152;           //  8,388,608
  float* BUF5 = BUF4 + 8388608;           //  4,194,304 (small)
  float* opmA  = BUF5;                    // 1,048,576 (aT[i][c][s])
  float* opmB  = opmA + 1048576;          // 1,048,576 (bT[j][e][s])
  float* NRM   = opmB + 1048576;          // 65,536
  float* CM    = NRM + 65536;             // 256
  float* LOG   = CM + 256;                // 524,288  (msa logits [8,256,256])
  float* GBIAS = LOG + 524288;            // 262,144  (grid bias [4,256,256])

  // live state = output buffers
  hipMemcpyAsync(msaO, d_in[IN_MSA], (size_t)SR * C_MSA * 4, hipMemcpyDeviceToDevice, stream);
  hipMemcpyAsync(pairO, d_in[IN_PAIR], (size_t)MR * C_PAIR * 4, hipMemcpyDeviceToDevice, stream);

  k_colmask<<<1, 256, 0, stream>>>(msa_mask, CM);
  k_normij<<<256, 256, 0, stream>>>(msa_mask, NRM);

  // ---------------- OuterProductMean: pair += opm(msa) ----------------
  k_ln<<<SR, 32, 0, stream>>>(msaO, P(P_OPM_LN_G), P(P_OPM_LN_B), BUF3, C_MSA);
  // projections stored s-major: aT[i][c][s] (batched over z=s, rows=i)
  gemm(stream, BUF3, P(P_OPM_LEFT_W), opmA, msa_mask, 256, 32, 64, 1, 128,
       64, 1, 0, (long long)256 * 64,
       32, 1, 0, 0,
       4096, 128, 0, 1, 1.f, EPI_ROWMASK, 1, 256);
  gemm(stream, BUF3, P(P_OPM_RIGHT_W), opmB, msa_mask, 256, 32, 64, 1, 128,
       64, 1, 0, (long long)256 * 64,
       32, 1, 0, 0,
       4096, 128, 0, 1, 1.f, EPI_ROWMASK, 1, 256);
  k_opm<<<dim3(16, 256), 256, 0, stream>>>(opmA, opmB, P(P_OPM_OUT_W), P(P_OPM_OUT_B), NRM, pairO);

  // ---------------- MSA pair-biased row attention ----------------
  k_ln<<<SR, 32, 0, stream>>>(msaO, P(P_MA_ACT_LN_G), P(P_MA_ACT_LN_B), BUF3, C_MSA);
  k_ln<<<MR, 32, 0, stream>>>(pairO, P(P_MA_PAIR_LN_G), P(P_MA_PAIR_LN_B), BUF2, C_PAIR);
  // logits[h,i,j] = pz @ pair_logits_w  (store transposed into [h][i*256+j])
  gemm(stream, BUF2, P(P_MA_PAIR_LOGITS_W), LOG, nullptr, MR, 8, C_PAIR, 1, 1,
       C_PAIR, 1, 0, 0, 8, 1, 0, 0, 1, MR, 0, 0, 1.f, EPI_STORE);
  k_softmax256<<<8 * N_RES, 32, 0, stream>>>(LOG, CM);
  // v = act @ v_w  -> BUF0 [s,i,64]
  gemm_rm(stream, BUF3, P(P_MA_V_W), BUF0, nullptr, SR, C_MSA, C_MSA, 1.f, EPI_STORE);
  // o[s,i,h,d] = sum_j wts[h,i,j] v[s,j,h,d]  (batched over y=h, z=s)
  gemm(stream, LOG, BUF0, BUF1, nullptr, N_RES, 8, N_RES, 8, N_SEQ,
       N_RES, 1, (long long)MR, 0,
       C_MSA, 1, 8, (long long)N_RES * C_MSA,
       C_MSA, 1, 8, (long long)N_RES * C_MSA, 1.f, EPI_STORE);
  gemm_rm(stream, BUF3, P(P_MA_GATING_W), BUF1, nullptr, SR, C_MSA, C_MSA, 1.f, EPI_SIGMUL);
  gemm_rm(stream, BUF1, P(P_MA_OUT_W), msaO, nullptr, SR, C_MSA, C_MSA, 1.f, EPI_ADD);

  // ---------------- MSA transition (SwiGLU) ----------------
  k_ln<<<SR, 32, 0, stream>>>(msaO, P(P_MT_LN_G), P(P_MT_LN_B), BUF3, C_MSA);
  gemm_rm(stream, BUF3, P(P_MT_W1), BUF0, nullptr, SR, 8 * C_MSA, C_MSA, 1.f, EPI_STORE);
  k_glu<<<(SR * 256 + 255) / 256, 256, 0, stream>>>(BUF0, BUF1, (long long)SR * 256, 256);
  gemm_rm(stream, BUF1, P(P_MT_W2), msaO, nullptr, SR, C_MSA, 4 * C_MSA, 1.f, EPI_ADD);

  // ---------------- Triangle multiplications ----------------
  // proj/gate stored channel-major so both einsum operands are k-contiguous:
  //  outgoing: T[ch][r1*256+r2];  incoming: S[ch][r2*256+r1] (batched z=r1 store)
  for (int tm = 0; tm < 2; ++tm) {
    const float* lg = P(tm == 0 ? P_TMO_LN_G : P_TMI_LN_G);
    const float* lb = P(tm == 0 ? P_TMO_LN_B : P_TMI_LN_B);
    const float* cg = P(tm == 0 ? P_TMO_CLN_G : P_TMI_CLN_G);
    const float* cb = P(tm == 0 ? P_TMO_CLN_B : P_TMI_CLN_B);
    const float* wp = P(tm == 0 ? P_TMO_PROJ_W : P_TMI_PROJ_W);
    const float* wg = P(tm == 0 ? P_TMO_GATE_W : P_TMI_GATE_W);
    const float* wo = P(tm == 0 ? P_TMO_OUT_W : P_TMI_OUT_W);
    const float* wgg = P(tm == 0 ? P_TMO_GATING_W : P_TMI_GATING_W);

    k_ln<<<MR, 32, 0, stream>>>(pairO, lg, lb, BUF2, C_PAIR);
    if (tm == 0) {
      // T[ch][m], m = r1*256+r2 (plain rows, transposed store)
      gemm(stream, BUF2, wp, BUF0, pair_mask, MR, 256, C_PAIR, 1, 1,
           C_PAIR, 1, 0, 0, 256, 1, 0, 0, 1, MR, 0, 0, 1.f, EPI_ROWMASK, 1, 0);
      gemm(stream, BUF2, wg, BUF0, nullptr, MR, 256, C_PAIR, 1, 1,
           C_PAIR, 1, 0, 0, 256, 1, 0, 0, 1, MR, 0, 0, 1.f, EPI_SIGMUL, 1, 0);
      // z[i,j,c] = sum_k a[i,k,c]*b[j,k,c] = sum_k T[c][i*256+k]*T[128+c][j*256+k]
      gemm(stream, BUF0, BUF0 + 128LL * MR, BUF1, nullptr, N_RES, N_RES, N_RES, C_PAIR, 1,
           256, 1, (long long)MR, 0,
           1, 256, (long long)MR, 0,
           (long long)N_RES * C_PAIR, C_PAIR, 1, 0, 1.f, EPI_STORE);
    } else {
      // S[ch][r2*256 + r1]: batched store (z=r1, rows m=r2)
      gemm(stream, BUF2, wp, BUF0, pair_mask, 256, 256, C_PAIR, 1, 256,
           C_PAIR, 1, 0, (long long)N_RES * C_PAIR,
           256, 1, 0, 0,
           256, MR, 0, 1, 1.f, EPI_ROWMASK, 1, 256);
      gemm(stream, BUF2, wg, BUF0, nullptr, 256, 256, C_PAIR, 1, 256,
           C_PAIR, 1, 0, (long long)N_RES * C_PAIR,
           256, 1, 0, 0,
           256, MR, 0, 1, 1.f, EPI_SIGMUL, 1, 256);
      // z[i,j,c] = sum_k a[k,j,c]*b[k,i,c] = sum_k S[128+c][i*256+k]*S[c][j*256+k]
      gemm(stream, BUF0 + 128LL * MR, BUF0, BUF1, nullptr, N_RES, N_RES, N_RES, C_PAIR, 1,
           256, 1, (long long)MR, 0,
           1, 256, (long long)MR, 0,
           (long long)N_RES * C_PAIR, C_PAIR, 1, 0, 1.f, EPI_STORE);
    }
    k_ln<<<MR, 32, 0, stream>>>(BUF1, cg, cb, BUF4, C_PAIR);
    gemm_rm(stream, BUF4, wo, BUF1, nullptr, MR, C_PAIR, C_PAIR, 1.f, EPI_STORE);
    gemm_rm(stream, BUF2, wgg, BUF1, nullptr, MR, C_PAIR, C_PAIR, 1.f, EPI_SIGMUL);
    k_add<<<(MR * C_PAIR + 255) / 256, 256, 0, stream>>>(pairO, BUF1, (long long)MR * C_PAIR);
  }

  // ---------------- Grid self-attentions ----------------
  const float qscale = 0.17677669529663687f;  // (32)^-0.5
  for (int ga = 0; ga < 2; ++ga) {
    int base = (ga == 0) ? P_PA1_BIAS_W : P_PA2_BIAS_W;
    const float* lg = P(ga == 0 ? P_PA1_LN_G : P_PA2_LN_G);
    const float* lb = P(ga == 0 ? P_PA1_LN_B : P_PA2_LN_B);
    k_ln<<<MR, 32, 0, stream>>>(pairO, lg, lb, BUF2, C_PAIR);
    // bias[h,q,k] (pre-transpose, both variants)
    gemm(stream, BUF2, P(base + 0), GBIAS, nullptr, MR, 4, C_PAIR, 1, 1,
         C_PAIR, 1, 0, 0, 4, 1, 0, 0, 1, MR, 0, 0, 1.f, EPI_STORE);
    float* Q = BUF0;
    float* K_ = BUF0 + 8388608;
    float* V = BUF4;
    gemm_rm(stream, BUF2, P(base + 1), Q, nullptr, MR, C_PAIR, C_PAIR, qscale, EPI_STORE);
    gemm_rm(stream, BUF2, P(base + 2), K_, nullptr, MR, C_PAIR, C_PAIR, 1.f, EPI_STORE);
    gemm_rm(stream, BUF2, P(base + 3), V, nullptr, MR, C_PAIR, C_PAIR, 1.f, EPI_STORE);
    long long sRow = (ga == 0) ? C_PAIR : (long long)N_RES * C_PAIR;
    long long sB = (ga == 0) ? (long long)N_RES * C_PAIR : C_PAIR;
    k_attn<<<dim3(16, 4, 256), 64, 0, stream>>>(Q, K_, V, GBIAS, pair_mask, BUF1,
                                                sRow, sB, sRow, sB);
    gemm_rm(stream, BUF2, P(base + 4), BUF1, nullptr, MR, C_PAIR, C_PAIR, 1.f, EPI_SIGMUL);
    gemm_rm(stream, BUF1, P(base + 5), pairO, nullptr, MR, C_PAIR, C_PAIR, 1.f, EPI_ADD);
  }

  // ---------------- Pair transition (chunked SwiGLU) ----------------
  k_ln<<<MR, 32, 0, stream>>>(pairO, P(P_PT_LN_G), P(P_PT_LN_B), BUF2, C_PAIR);
  const int CH = 16384;
  for (int c = 0; c < MR / CH; ++c) {
    const float* xin = BUF2 + (long long)c * CH * C_PAIR;
    float* pout = pairO + (long long)c * CH * C_PAIR;
    gemm_rm(stream, xin, P(P_PT_W1), BUF0, nullptr, CH, 8 * C_PAIR, C_PAIR, 1.f, EPI_STORE);
    k_glu<<<((long long)CH * 512 + 255) / 256, 256, 0, stream>>>(BUF0, BUF1, (long long)CH * 512, 512);
    gemm_rm(stream, BUF1, P(P_PT_W2), pout, nullptr, CH, C_PAIR, 4 * C_PAIR, 1.f, EPI_ADD);
  }
}
